// M3GNetInteraction_34514357191193
// MI455X (gfx1250) — matrix operand order repaired and checked by reading, not compiled, not run on hardware
//
#include <hip/hip_runtime.h>

#define N_ATOMS   20000
#define N_EDGES   640000
#define N_TRIP    2000000
#define CH        128
#define EF        64
#define CUTOFF_F  5.0f
#define PI_F      3.14159265358979323846f

typedef __attribute__((ext_vector_type(16))) _Float16 v16h;
typedef __attribute__((ext_vector_type(8)))  float    v8f;

__device__ __forceinline__ v8f wmma_f16(v16h a, v16h b, v8f c) {
  // D = A(16x32 f16) * B(32x16 f16) + C(16x16 f32)
  return __builtin_amdgcn_wmma_f32_16x16x32_f16(false, a, false, b, (short)0, c, false, false);
}

__device__ __forceinline__ float silu_f(float x) { return x / (1.f + __expf(-x)); }
__device__ __forceinline__ float sigm_f(float x) { return 1.f / (1.f + __expf(-x)); }

// ---- A fragment (16x32, f16) from row-major f32 [*, ld]
// ISA 7.12.2: lanes 0-15 -> M=lane, K = {k0+0..7, k0+16..23}; lanes 16-31 -> K = {k0+8..15, k0+24..31}
__device__ __forceinline__ v16h load_A_f32(const float* __restrict__ A, int ld, int m0, int k0, int lane) {
  int m = m0 + (lane & 15);
  const float* p = A + (size_t)m * ld + k0 + ((lane >> 4) << 3);
  v16h a;
#pragma unroll
  for (int i = 0; i < 8; ++i) {
    a[i]     = (_Float16)p[i];
    a[i + 8] = (_Float16)p[i + 16];
  }
  return a;
}

// ---- A fragment from LDS f16 row-major [16 x ld]
__device__ __forceinline__ v16h load_A_lds(const _Float16* L, int ld, int k0, int lane) {
  int m = lane & 15;
  const _Float16* p = L + m * ld + k0 + ((lane >> 4) << 3);
  v16h a;
#pragma unroll
  for (int i = 0; i < 8; ++i) {
    a[i]     = p[i];
    a[i + 8] = p[i + 16];
  }
  return a;
}

// ---- B fragment from pre-packed fragment table: one 32B vector load per lane.
// Layout: frag[((kc*NT + nt)*32 + lane)*16 + e]
__device__ __forceinline__ v16h load_B_frag(const _Float16* __restrict__ F, int NT, int kc, int nt, int lane) {
  return *reinterpret_cast<const v16h*>(F + ((((kc * NT) + nt) * 32 + lane) << 4));
}

// ======================= weight repacking: W[K x N] f32 -> B-fragment f16 table
// k = kc*32 + (lane>>4)*8 + (e&7) + (e>>3)*16 ; n = nt*16 + (lane&15) ; zero-pad k >= Kreal
__global__ void pack_B_frag(const float* __restrict__ W, _Float16* __restrict__ out,
                            int Kreal, int N, int total) {
  int i = blockIdx.x * blockDim.x + threadIdx.x;
  if (i >= total) return;
  int e    = i & 15;
  int lane = (i >> 4) & 31;
  int tile = i >> 9;           // kc*NT + nt
  int NT   = N >> 4;
  int kc   = tile / NT;
  int nt   = tile - kc * NT;
  int k = kc * 32 + ((lane >> 4) << 3) + (e & 7) + ((e >> 3) << 4);
  int n = nt * 16 + (lane & 15);
  out[i] = (k < Kreal) ? (_Float16)W[(size_t)k * N + n] : (_Float16)0.f;
}

// ======================= dense GEMM: C[MxN] = A[MxK f32] @ Bfrag (f16 WMMA, f32 accum)
__global__ void __launch_bounds__(128)
gemm128_wmma(const float* __restrict__ A, const _Float16* __restrict__ Bf,
             float* __restrict__ C, int M, int N, int K) {
  int lane = threadIdx.x & 31;
  int tile = blockIdx.x * 4 + (threadIdx.x >> 5);   // one wave = one 16x16 tile
  int NT = N >> 4;
  int tm = (tile / NT) << 4;
  int tn4 = tile % NT;
  if (tm >= M) return;
  v8f acc = {};
  for (int kc = 0; kc < (K >> 5); ++kc) {
    v16h a = load_A_f32(A, K, tm, kc * 32, lane);
    v16h b = load_B_frag(Bf, NT, kc, tn4, lane);
    acc = wmma_f16(a, b, acc);
  }
  int n  = (tn4 << 4) + (lane & 15);
  int mb = tm + ((lane >> 4) << 3);
#pragma unroll
  for (int r = 0; r < 8; ++r) C[(size_t)(mb + r) * N + n] = acc[r];
}

// ======================= two-body: 16 edges per wave
__global__ void __launch_bounds__(128)
edge_two_body(const float* __restrict__ dist, const int* __restrict__ nbr,
              const float* __restrict__ h,
              const _Float16* __restrict__ W1f, const _Float16* __restrict__ W2f,
              const _Float16* __restrict__ G1f, const _Float16* __restrict__ G2f,
              float* __restrict__ h_acc) {
  __shared__ _Float16 lds_t1[4][16 * EF];
  int lane = threadIdx.x & 31;
  int wv   = threadIdx.x >> 5;
  _Float16* t1 = lds_t1[wv];
  int e0 = (blockIdx.x * 4 + wv) << 4;
  int m  = lane & 15;
  int kg = lane >> 4;

  // radial basis built directly in A-fragment layout (K = expansion index)
  float d   = dist[e0 + m];
  float cut = (d < CUTOFF_F) ? 0.5f * (1.f + __cosf(PI_F * d * (1.f / CUTOFF_F))) : 0.f;
  const float step   = CUTOFF_F / (float)(EF - 1);
  const float inv2s2 = (float)(EF * EF) / (2.f * CUTOFF_F * CUTOFF_F);

  v16h rbA[2];
#pragma unroll
  for (int kc = 0; kc < 2; ++kc) {
#pragma unroll
    for (int i = 0; i < 8; ++i) {
      float k1 = (float)(kc * 32 + kg * 8 + i);
      float k2 = k1 + 16.f;
      float x1 = d - k1 * step;
      float x2 = d - k2 * step;
      rbA[kc][i]     = (_Float16)(__expf(-x1 * x1 * inv2s2) * cut);
      rbA[kc][i + 8] = (_Float16)(__expf(-x2 * x2 * inv2s2) * cut);
    }
  }

  // ---- gate stage 1: g1 = silu(rb @ G1) -> LDS
#pragma unroll
  for (int nt = 0; nt < 4; ++nt) {
    v8f acc = {};
#pragma unroll
    for (int kc = 0; kc < 2; ++kc)
      acc = wmma_f16(rbA[kc], load_B_frag(G1f, 4, kc, nt, lane), acc);
    int col = nt * 16 + m;
#pragma unroll
    for (int r = 0; r < 8; ++r)
      t1[(kg * 8 + r) * EF + col] = (_Float16)silu_f(acc[r]);
  }
  __syncthreads();
  v16h gA[2];
  gA[0] = load_A_lds(t1, EF, 0, lane);
  gA[1] = load_A_lds(t1, EF, 32, lane);

  // gate stage 2: C-layout == later read pattern -> keep in registers
  float gatev[8][8];
#pragma unroll
  for (int nt = 0; nt < 8; ++nt) {
    v8f acc = {};
#pragma unroll
    for (int kc = 0; kc < 2; ++kc)
      acc = wmma_f16(gA[kc], load_B_frag(G2f, 8, kc, nt, lane), acc);
#pragma unroll
    for (int r = 0; r < 8; ++r) gatev[nt][r] = sigm_f(acc[r]);
  }
  __syncthreads();

  // ---- main stage 1: t1 = silu(rb @ W1) -> LDS
#pragma unroll
  for (int nt = 0; nt < 4; ++nt) {
    v8f acc = {};
#pragma unroll
    for (int kc = 0; kc < 2; ++kc)
      acc = wmma_f16(rbA[kc], load_B_frag(W1f, 4, kc, nt, lane), acc);
    int col = nt * 16 + m;
#pragma unroll
    for (int r = 0; r < 8; ++r)
      t1[(kg * 8 + r) * EF + col] = (_Float16)silu_f(acc[r]);
  }
  __syncthreads();
  v16h tA[2];
  tA[0] = load_A_lds(t1, EF, 0, lane);
  tA[1] = load_A_lds(t1, EF, 32, lane);

  int i0[8], j1[8];
#pragma unroll
  for (int r = 0; r < 8; ++r) {
    int e = e0 + kg * 8 + r;
    i0[r] = nbr[e];              // neighbour_list[0]
    j1[r] = nbr[N_EDGES + e];    // neighbour_list[1]
  }

  // main stage 2 + gate + gather h + fused segment-sum scatter (L2-resident atomics)
#pragma unroll
  for (int nt = 0; nt < 8; ++nt) {
    v8f acc = {};
#pragma unroll
    for (int kc = 0; kc < 2; ++kc)
      acc = wmma_f16(tA[kc], load_B_frag(W2f, 8, kc, nt, lane), acc);
    int col = nt * 16 + m;
#pragma unroll
    for (int r = 0; r < 8; ++r) {
      float v = acc[r] * gatev[nt][r] * h[(size_t)j1[r] * CH + col];
      atomicAdd(&h_acc[(size_t)i0[r] * CH + col], v);
    }
  }
}

// ======================= three-body: 16 triplets per wave (K=3 zero-padded in packed W1/G1)
__global__ void __launch_bounds__(128)
triplet_three_body(const float* __restrict__ ang, const float* __restrict__ rij,
                   const float* __restrict__ rik, const int* __restrict__ trip,
                   const float* __restrict__ h,
                   const _Float16* __restrict__ W1f, const _Float16* __restrict__ W2f,
                   const _Float16* __restrict__ G1f, const _Float16* __restrict__ G2f,
                   float* __restrict__ T) {
  __shared__ _Float16 lds_t1[4][16 * EF];
  int lane = threadIdx.x & 31;
  int wv   = threadIdx.x >> 5;
  _Float16* t1 = lds_t1[wv];
  int t0 = (blockIdx.x * 4 + wv) << 4;
  int m  = lane & 15;
  int kg = lane >> 4;
  int t  = t0 + m;

  // A-fragment of [r_ij, r_ik, cos(angle), 0...]: K<3 lives only in lane-half 0, elems 0..2
  v16h fA = {};
  if (kg == 0) {
    fA[0] = (_Float16)rij[t];
    fA[1] = (_Float16)rik[t];
    fA[2] = (_Float16)__cosf(ang[t]);
  }

  // ---- gate stage 1: silu(f3 @ G1) -> LDS  (single K chunk)
#pragma unroll
  for (int nt = 0; nt < 4; ++nt) {
    v8f acc = {};
    acc = wmma_f16(fA, load_B_frag(G1f, 4, 0, nt, lane), acc);
    int col = nt * 16 + m;
#pragma unroll
    for (int r = 0; r < 8; ++r)
      t1[(kg * 8 + r) * EF + col] = (_Float16)silu_f(acc[r]);
  }
  __syncthreads();
  v16h gA[2];
  gA[0] = load_A_lds(t1, EF, 0, lane);
  gA[1] = load_A_lds(t1, EF, 32, lane);

  float gatev[8][8];
#pragma unroll
  for (int nt = 0; nt < 8; ++nt) {
    v8f acc = {};
#pragma unroll
    for (int kc = 0; kc < 2; ++kc)
      acc = wmma_f16(gA[kc], load_B_frag(G2f, 8, kc, nt, lane), acc);
#pragma unroll
    for (int r = 0; r < 8; ++r) gatev[nt][r] = sigm_f(acc[r]);
  }
  __syncthreads();

  // ---- main stage 1: silu(f3 @ W1) -> LDS
#pragma unroll
  for (int nt = 0; nt < 4; ++nt) {
    v8f acc = {};
    acc = wmma_f16(fA, load_B_frag(W1f, 4, 0, nt, lane), acc);
    int col = nt * 16 + m;
#pragma unroll
    for (int r = 0; r < 8; ++r)
      t1[(kg * 8 + r) * EF + col] = (_Float16)silu_f(acc[r]);
  }
  __syncthreads();
  v16h tA[2];
  tA[0] = load_A_lds(t1, EF, 0, lane);
  tA[1] = load_A_lds(t1, EF, 32, lane);

  int idx1[8];
#pragma unroll
  for (int r = 0; r < 8; ++r)
    idx1[r] = trip[(size_t)(t0 + kg * 8 + r) * 3 + 1];

#pragma unroll
  for (int nt = 0; nt < 8; ++nt) {
    v8f acc = {};
#pragma unroll
    for (int kc = 0; kc < 2; ++kc)
      acc = wmma_f16(tA[kc], load_B_frag(W2f, 8, kc, nt, lane), acc);
    int col = nt * 16 + m;
#pragma unroll
    for (int r = 0; r < 8; ++r) {
      float v = acc[r] * gatev[nt][r] * h[(size_t)idx1[r] * CH + col];
      atomicAdd(&T[(size_t)idx1[r] * CH + col], v);
    }
  }
}

// ======================= helpers
__global__ void zero_f32(float* __restrict__ p, int n) {
  int i = blockIdx.x * blockDim.x + threadIdx.x;
  if (i < n) p[i] = 0.f;
}

// three_body rows e < N_ATOMS equal T[e]; fold into the edge segment-sum
__global__ void scatter_T(const float* __restrict__ T, const int* __restrict__ nbr,
                          float* __restrict__ h_acc) {
  int i = blockIdx.x * blockDim.x + threadIdx.x;  // over N_ATOMS*CH
  int e = i >> 7;
  int c = i & (CH - 1);
  atomicAdd(&h_acc[(size_t)nbr[e] * CH + c], T[i]);
}

extern "C" void kernel_launch(void* const* d_in, const int* in_sizes, int n_in,
                              void* d_out, int out_size, void* d_ws, size_t ws_size,
                              hipStream_t stream) {
  const float* features = (const float*)d_in[0];
  const float* dist     = (const float*)d_in[1];
  const float* angles   = (const float*)d_in[2];
  const float* r_ij     = (const float*)d_in[3];
  const float* r_ik     = (const float*)d_in[4];
  const int*   nbr      = (const int*)d_in[5];
  const int*   trip     = (const int*)d_in[6];
  const float* W_pre    = (const float*)d_in[7];
  const float* tb_W1    = (const float*)d_in[8];
  const float* tb_W2    = (const float*)d_in[9];
  const float* tb_G1    = (const float*)d_in[10];
  const float* tb_G2    = (const float*)d_in[11];
  const float* th_W1    = (const float*)d_in[12];
  const float* th_W2    = (const float*)d_in[13];
  const float* th_G1    = (const float*)d_in[14];
  const float* th_G2    = (const float*)d_in[15];
  const float* W_post   = (const float*)d_in[16];
  float* out = (float*)d_out;

  float* ws    = (float*)d_ws;
  float* h     = ws;                              // N_ATOMS*CH f32
  float* h_acc = ws + (size_t)N_ATOMS * CH;       // N_ATOMS*CH f32
  float* T     = ws + 2 * (size_t)N_ATOMS * CH;   // N_ATOMS*CH f32

  // packed f16 B-fragment tables (32B-aligned: 30.72MB float region is 32B multiple)
  _Float16* fp = (_Float16*)(ws + 3 * (size_t)N_ATOMS * CH);
  _Float16* pWpre  = fp;                 // 128x128 -> 16384
  _Float16* pWpost = fp + 16384;         // 16384
  _Float16* pTbW1  = fp + 32768;         // 64x64   -> 4096
  _Float16* pTbG1  = fp + 36864;         // 4096
  _Float16* pTbW2  = fp + 40960;         // 64x128  -> 8192
  _Float16* pTbG2  = fp + 49152;         // 8192
  _Float16* pThW1  = fp + 57344;         // 3->pad32 x64 -> 2048
  _Float16* pThG1  = fp + 59392;         // 2048
  _Float16* pThW2  = fp + 61440;         // 8192
  _Float16* pThG2  = fp + 69632;         // 8192

  // ---- pack weights into fragment layout (tiny, branch-free hot loops afterwards)
  pack_B_frag<<<64, 256, 0, stream>>>(W_pre,  pWpre, 128, 128, 16384);
  pack_B_frag<<<64, 256, 0, stream>>>(W_post, pWpost, 128, 128, 16384);
  pack_B_frag<<<16, 256, 0, stream>>>(tb_W1,  pTbW1,  64,  64,  4096);
  pack_B_frag<<<16, 256, 0, stream>>>(tb_G1,  pTbG1,  64,  64,  4096);
  pack_B_frag<<<32, 256, 0, stream>>>(tb_W2,  pTbW2,  64, 128,  8192);
  pack_B_frag<<<32, 256, 0, stream>>>(tb_G2,  pTbG2,  64, 128,  8192);
  pack_B_frag<<< 8, 256, 0, stream>>>(th_W1,  pThW1,   3,  64,  2048);
  pack_B_frag<<< 8, 256, 0, stream>>>(th_G1,  pThG1,   3,  64,  2048);
  pack_B_frag<<<32, 256, 0, stream>>>(th_W2,  pThW2,  64, 128,  8192);
  pack_B_frag<<<32, 256, 0, stream>>>(th_G2,  pThG2,  64, 128,  8192);

  // zero accumulators (h_acc, T contiguous)
  zero_f32<<<(2 * N_ATOMS * CH) / 256, 256, 0, stream>>>(h_acc, 2 * N_ATOMS * CH);

  // h = features @ W_pre
  gemm128_wmma<<<(N_ATOMS / 16) * (CH / 16) / 4, 128, 0, stream>>>(
      features, pWpre, h, N_ATOMS, CH, CH);

  // two-body fused into h_acc
  edge_two_body<<<N_EDGES / 64, 128, 0, stream>>>(
      dist, nbr, h, pTbW1, pTbW2, pTbG1, pTbG2, h_acc);

  // three-body table T
  triplet_three_body<<<N_TRIP / 64, 128, 0, stream>>>(
      angles, r_ij, r_ik, trip, h, pThW1, pThW2, pThG1, pThG2, T);

  // fold T into h_acc via nbr0
  scatter_T<<<(N_ATOMS * CH) / 256, 256, 0, stream>>>(T, nbr, h_acc);

  // out = h_acc @ W_post
  gemm128_wmma<<<(N_ATOMS / 16) * (CH / 16) / 4, 128, 0, stream>>>(
      h_acc, pWpost, out, N_ATOMS, CH, CH);
}
